// TransformerBlock_781684047898
// MI455X (gfx1250) — matrix operand order repaired and checked by reading, not compiled, not run on hardware
//
#include <hip/hip_runtime.h>

// ---------------------------------------------------------------------------
// Types for CDNA5 WMMA (gfx1250, wave32)
// ---------------------------------------------------------------------------
typedef __attribute__((ext_vector_type(16))) __bf16 v16bf;
typedef __attribute__((ext_vector_type(8)))  float  v8f;

struct B32x2 { uint4 lo, hi; };   // 32 bytes = one 16x32 bf16 fragment per lane

__device__ __forceinline__ v8f wmma_bf16(const B32x2& a, const B32x2& b, v8f c) {
    return __builtin_amdgcn_wmma_f32_16x16x32_bf16(
        false, __builtin_bit_cast(v16bf, a),
        false, __builtin_bit_cast(v16bf, b),
        (short)0, c, false, false);
}

__device__ __forceinline__ unsigned short f2bf(float f) {
    union { float f; unsigned u; } x; x.f = f;
    unsigned r = x.u + 0x7FFFu + ((x.u >> 16) & 1u);   // round-to-nearest-even
    return (unsigned short)(r >> 16);
}

__device__ __forceinline__ float silu(float v) {
    return v / (1.0f + __expf(-v));
}

// CDNA5 async copy: global -> LDS, 16 bytes per lane, tracked by ASYNCcnt.
// LDS generic pointers carry the wave-relative LDS offset in their low 32 bits
// (the flat->LDS aperture uses the low 32 bits as the LDS address), so truncation gives VDST.
__device__ __forceinline__ void async_g2l_b128(const void* gsrc, void* ldst) {
    unsigned loff = (unsigned)(uintptr_t)ldst;
    asm volatile("global_load_async_to_lds_b128 %0, %1, off"
                 :: "v"(loff), "v"(gsrc) : "memory");
}
__device__ __forceinline__ void wait_asynccnt0() {
    asm volatile("s_wait_asynccnt 0x0" ::: "memory");
}

static constexpr int S_ = 2048, D_ = 1024, L_ = 256, H_ = 1024, NH_ = 16, HD_ = 64;

// ---------------------------------------------------------------------------
// f32 -> bf16 convert (weights), 4 elements/thread
// ---------------------------------------------------------------------------
__global__ void __launch_bounds__(256) cvt_bf16x4_kernel(const float* __restrict__ in,
                                                         unsigned short* __restrict__ out,
                                                         int n4) {
    int i = blockIdx.x * 256 + threadIdx.x;
    if (i < n4) {
        float4 f = ((const float4*)in)[i];
        unsigned short r[4] = { f2bf(f.x), f2bf(f.y), f2bf(f.z), f2bf(f.w) };
        ((uint2*)out)[i] = *(const uint2*)r;
    }
}

// ---------------------------------------------------------------------------
// RMSNorm: f32 [rows][D] * g -> bf16 [rows][D]
// ---------------------------------------------------------------------------
__global__ void __launch_bounds__(256) rmsnorm_bf16_kernel(const float* __restrict__ x,
                                                           const float* __restrict__ g,
                                                           unsigned short* __restrict__ out,
                                                           int d) {
    __shared__ float red[8];
    const int row = blockIdx.x;
    const float* xr = x + (size_t)row * d;
    float ss = 0.f;
    for (int i = threadIdx.x; i < d; i += 256) { float v = xr[i]; ss += v * v; }
    #pragma unroll
    for (int m = 16; m >= 1; m >>= 1) ss += __shfl_xor(ss, m, 32);
    if ((threadIdx.x & 31) == 0) red[threadIdx.x >> 5] = ss;
    __syncthreads();
    if (threadIdx.x < 8) {
        float t = red[threadIdx.x];
        #pragma unroll
        for (int m = 4; m >= 1; m >>= 1) t += __shfl_xor(t, m, 8);
        if (threadIdx.x == 0) red[0] = t;
    }
    __syncthreads();
    const float rinv = rsqrtf(red[0] / (float)d + 1e-6f);
    for (int i = threadIdx.x; i < d; i += 256)
        out[(size_t)row * d + i] = f2bf(xr[i] * rinv * g[i]);
}

// ---------------------------------------------------------------------------
// Tiled bf16 WMMA GEMM:  out = epilogue(A[M,K] @ B[K,N] + bias, res)
// BM=64, BN=128, BK=64. 256 threads = 8 waves, 2x4 wave grid, 32x32 per wave.
// A tile staged via GLOBAL_LOAD_ASYNC_TO_LDS_B128 (ASYNCcnt); B transposed
// through VGPRs into LDS. 8 wmma per wave per stage.
// MODE: 0 = bf16 row-major      1 = bf16 transposed ([N][M])
//       2 = f32 + residual      3 = silu -> bf16     4 = silu + residual -> f32
// ---------------------------------------------------------------------------
#define BM 64
#define BN 128
#define BK 64

template <int MODE>
__global__ void __launch_bounds__(256)
gemm_bf16_kernel(const unsigned short* __restrict__ A,
                 const unsigned short* __restrict__ Bm,
                 const float* __restrict__ bias,
                 const float* __restrict__ res,
                 void* __restrict__ out,
                 int M, int N, int K) {
    __shared__ unsigned short sA[BM][BK];        // 8 KB
    __shared__ unsigned short sBT[BN][BK];       // 16 KB, B transposed: [n][k]

    const int tid   = threadIdx.x;
    const int wave  = tid >> 5;
    const int lane  = tid & 31;
    const int row16 = lane & 15;
    const int sel   = lane >> 4;                 // half-wave select
    const int bm0 = blockIdx.y * BM, bn0 = blockIdx.x * BN;
    const int wm0 = (wave >> 2) * 32;            // wave tile origin in block
    const int wn0 = (wave & 3) * 32;

    v8f acc[2][2] = {};

    for (int k0 = 0; k0 < K; k0 += BK) {
        // --- stage A tile 64x64: async global->LDS, 2 x b128 per thread ---
        #pragma unroll
        for (int i = 0; i < 2; ++i) {
            const int c = tid + i * 256;                     // 512 chunks of 8
            const int r = c >> 3, col = (c & 7) * 8;
            async_g2l_b128(&A[(size_t)(bm0 + r) * K + k0 + col], &sA[r][col]);
        }
        // --- stage B tile transposed: 64x128 -> sBT[n][k] ---
        #pragma unroll
        for (int i = 0; i < 4; ++i) {
            const int c = tid + i * 256;                     // 1024 chunks of 8
            const int kk = c >> 4, nch = (c & 15) * 8;
            uint4 d = *(const uint4*)&Bm[(size_t)(k0 + kk) * N + bn0 + nch];
            const unsigned short* sp = (const unsigned short*)&d;
            #pragma unroll
            for (int j = 0; j < 8; ++j) sBT[nch + j][kk] = sp[j];
        }
        // --- prefetch next K tiles into cache while we compute this one ---
        if (k0 + BK < K) {
            __builtin_prefetch(&A[(size_t)(bm0 + (tid >> 2)) * K + k0 + BK], 0, 1);
            __builtin_prefetch(&Bm[(size_t)(k0 + BK + (tid >> 2)) * N + bn0 + (tid & 3) * 32], 0, 1);
        }
        wait_asynccnt0();
        __syncthreads();

        // --- fragments (ISA 16-bit A 16x32 / B 32x16 lane layouts) ---
        B32x2 af[2][2], bfr[2][2];
        #pragma unroll
        for (int i = 0; i < 2; ++i) {
            const int r = wm0 + i * 16 + row16;
            #pragma unroll
            for (int kc = 0; kc < 2; ++kc) {
                af[i][kc].lo = *(const uint4*)&sA[r][kc * 32 + sel * 8];
                af[i][kc].hi = *(const uint4*)&sA[r][kc * 32 + sel * 8 + 16];
            }
        }
        #pragma unroll
        for (int j = 0; j < 2; ++j) {
            const int c = wn0 + j * 16 + row16;
            #pragma unroll
            for (int kc = 0; kc < 2; ++kc) {
                bfr[j][kc].lo = *(const uint4*)&sBT[c][kc * 32 + sel * 8];
                bfr[j][kc].hi = *(const uint4*)&sBT[c][kc * 32 + sel * 8 + 16];
            }
        }
        #pragma unroll
        for (int i = 0; i < 2; ++i)
            #pragma unroll
            for (int j = 0; j < 2; ++j) {
                acc[i][j] = wmma_bf16(af[i][0], bfr[j][0], acc[i][j]);
                acc[i][j] = wmma_bf16(af[i][1], bfr[j][1], acc[i][j]);
            }
        __syncthreads();
    }

    // --- epilogue (C layout: VGPR r -> row r (lanes 0-15) / row r+8 (16-31)) ---
    #pragma unroll
    for (int i = 0; i < 2; ++i) {
        #pragma unroll
        for (int j = 0; j < 2; ++j) {
            #pragma unroll
            for (int r = 0; r < 8; ++r) {
                const int grow = bm0 + wm0 + i * 16 + sel * 8 + r;
                const int gcol = bn0 + wn0 + j * 16 + row16;
                float v = acc[i][j][r] + bias[gcol];
                if constexpr (MODE == 0) {
                    ((unsigned short*)out)[(size_t)grow * N + gcol] = f2bf(v);
                } else if constexpr (MODE == 1) {
                    ((unsigned short*)out)[(size_t)gcol * M + grow] = f2bf(v);
                } else if constexpr (MODE == 2) {
                    ((float*)out)[(size_t)grow * N + gcol] =
                        v + res[(size_t)grow * N + gcol];
                } else if constexpr (MODE == 3) {
                    ((unsigned short*)out)[(size_t)grow * N + gcol] = f2bf(silu(v));
                } else {
                    ((float*)out)[(size_t)grow * N + gcol] =
                        silu(v) + res[(size_t)grow * N + gcol];
                }
            }
        }
    }
}

// ---------------------------------------------------------------------------
// Flash attention (causal, HD=64): one wave per (head, 16-query tile).
// q,k bf16 row-major [S][H]; v bf16 transposed [H][S]; out bf16 [S][H].
// Streams 32 keys per iteration: 4 wmma for QK^T, 4 wmma for P*V.
// ---------------------------------------------------------------------------
__global__ void __launch_bounds__(128)
mla_attention_kernel(const unsigned short* __restrict__ qb,
                     const unsigned short* __restrict__ kb,
                     const unsigned short* __restrict__ vtb,
                     unsigned short* __restrict__ ob) {
    __shared__ unsigned short pLds[4][16][32];   // per-wave P staging, 4 KB

    const int wave = threadIdx.x >> 5;
    const int lane = threadIdx.x & 31;
    const int row16 = lane & 15;
    const int sel   = lane >> 4;
    const int task = blockIdx.x * 4 + wave;      // NH * S/16 = 2048 tasks
    const int head = task >> 7;                  // / (S/16)
    const int m0 = (task & 127) * 16;

    // Q fragments: two K=32 chunks across HD=64
    B32x2 qf[2];
    {
        const unsigned short* qrow = qb + (size_t)(m0 + row16) * H_ + head * HD_;
        qf[0].lo = *(const uint4*)&qrow[sel * 8];
        qf[0].hi = *(const uint4*)&qrow[sel * 8 + 16];
        qf[1].lo = *(const uint4*)&qrow[32 + sel * 8];
        qf[1].hi = *(const uint4*)&qrow[32 + sel * 8 + 16];
    }

    v8f accO[4] = {};
    float mrow[8], lrow[8];
    #pragma unroll
    for (int r = 0; r < 8; ++r) { mrow[r] = -1e30f; lrow[r] = 0.f; }

    const int nkv = m0 + 16;                     // causal horizon
    for (int j0 = 0; j0 < nkv; j0 += 32) {
        // prefetch next KV block while computing this one
        if (j0 + 32 < nkv) {
            __builtin_prefetch(&kb[(size_t)(j0 + 32 + lane) * H_ + head * HD_], 0, 1);
            __builtin_prefetch(&vtb[(size_t)(head * HD_ + (lane & 15) * 4) * S_ + j0 + 32], 0, 1);
        }
        // --- scores: two 16x16 tiles over 32 keys ---
        v8f s[2] = {};
        #pragma unroll
        for (int t = 0; t < 2; ++t) {
            #pragma unroll
            for (int c = 0; c < 2; ++c) {
                const unsigned short* krow =
                    kb + (size_t)(j0 + t * 16 + row16) * H_ + head * HD_ + c * 32 + sel * 8;
                B32x2 bfr;
                bfr.lo = *(const uint4*)&krow[0];
                bfr.hi = *(const uint4*)&krow[16];
                s[t] = wmma_bf16(qf[c], bfr, s[t]);
            }
        }
        // --- online softmax on C-fragment layout ---
        #pragma unroll
        for (int r = 0; r < 8; ++r) {
            const int rowg = m0 + sel * 8 + r;
            float v0 = s[0][r] * 0.125f;                     // 1/sqrt(64)
            float v1 = s[1][r] * 0.125f;
            if (j0 + row16      > rowg) v0 = -1e30f;
            if (j0 + 16 + row16 > rowg) v1 = -1e30f;
            float mx = fmaxf(v0, v1);
            #pragma unroll
            for (int msk = 8; msk >= 1; msk >>= 1) mx = fmaxf(mx, __shfl_xor(mx, msk, 16));
            const float mnew  = fmaxf(mrow[r], mx);
            const float scale = __expf(mrow[r] - mnew);
            const float p0 = __expf(v0 - mnew);
            const float p1 = __expf(v1 - mnew);
            float ps = p0 + p1;
            #pragma unroll
            for (int msk = 8; msk >= 1; msk >>= 1) ps += __shfl_xor(ps, msk, 16);
            lrow[r] = lrow[r] * scale + ps;
            mrow[r] = mnew;
            #pragma unroll
            for (int nt = 0; nt < 4; ++nt) accO[nt][r] *= scale;
            pLds[wave][sel * 8 + r][row16]      = f2bf(p0);
            pLds[wave][sel * 8 + r][16 + row16] = f2bf(p1);
        }
        // --- re-read P in A-fragment layout (wave-private; DS ops in-order) ---
        B32x2 pf;
        pf.lo = *(const uint4*)&pLds[wave][row16][sel * 8];
        pf.hi = *(const uint4*)&pLds[wave][row16][sel * 8 + 16];
        // --- P (16x32) @ V (32x64): V transposed so columns are contiguous ---
        #pragma unroll
        for (int nt = 0; nt < 4; ++nt) {
            const unsigned short* vcol =
                vtb + (size_t)(head * HD_ + nt * 16 + row16) * S_ + j0 + sel * 8;
            B32x2 bfr;
            bfr.lo = *(const uint4*)&vcol[0];
            bfr.hi = *(const uint4*)&vcol[16];
            accO[nt] = wmma_bf16(pf, bfr, accO[nt]);
        }
    }
    // --- normalize + store bf16 ---
    #pragma unroll
    for (int nt = 0; nt < 4; ++nt) {
        #pragma unroll
        for (int r = 0; r < 8; ++r) {
            const int rowg = m0 + sel * 8 + r;
            const int colg = head * HD_ + nt * 16 + row16;
            ob[(size_t)rowg * H_ + colg] = f2bf(accO[nt][r] / lrow[r]);
        }
    }
}

// ---------------------------------------------------------------------------
// Host-side orchestration
// ---------------------------------------------------------------------------
extern "C" void kernel_launch(void* const* d_in, const int* in_sizes, int n_in,
                              void* d_out, int out_size, void* d_ws, size_t ws_size,
                              hipStream_t stream) {
    const float* x     = (const float*)d_in[0];
    const float* g1    = (const float*)d_in[1];
    const float* Wq    = (const float*)d_in[2];
    const float* bq    = (const float*)d_in[3];
    const float* Wdown = (const float*)d_in[4];
    const float* bdown = (const float*)d_in[5];
    const float* Wk    = (const float*)d_in[6];
    const float* bk    = (const float*)d_in[7];
    const float* Wv    = (const float*)d_in[8];
    const float* bv    = (const float*)d_in[9];
    const float* Wo    = (const float*)d_in[10];
    const float* bo    = (const float*)d_in[11];
    const float* g2    = (const float*)d_in[12];
    const float* W1    = (const float*)d_in[13];
    const float* b1    = (const float*)d_in[14];
    const float* W2    = (const float*)d_in[15];
    const float* b2    = (const float*)d_in[16];

    size_t off = 0;
    auto wsAlloc = [&](size_t bytes) -> void* {
        void* p = (char*)d_ws + off;
        off += (bytes + 255) & ~(size_t)255;
        return p;
    };
    unsigned short* hb   = (unsigned short*)wsAlloc((size_t)S_ * D_ * 2);     // rmsnorm(x)
    unsigned short* qb   = (unsigned short*)wsAlloc((size_t)S_ * H_ * 2);
    unsigned short* cb   = (unsigned short*)wsAlloc((size_t)S_ * L_ * 2);
    unsigned short* kbuf = (unsigned short*)wsAlloc((size_t)S_ * H_ * 2);
    unsigned short* vtb  = (unsigned short*)wsAlloc((size_t)H_ * S_ * 2);     // transposed
    unsigned short* ob   = (unsigned short*)wsAlloc((size_t)S_ * H_ * 2);
    unsigned short* h2b  = (unsigned short*)wsAlloc((size_t)S_ * D_ * 2);
    unsigned short* f1b  = (unsigned short*)wsAlloc((size_t)S_ * 4 * D_ * 2);
    float*          st32 = (float*)wsAlloc((size_t)S_ * D_ * 4);              // stream
    unsigned short* wqb  = (unsigned short*)wsAlloc((size_t)D_ * H_ * 2);
    unsigned short* wdb  = (unsigned short*)wsAlloc((size_t)D_ * L_ * 2);
    unsigned short* wkb  = (unsigned short*)wsAlloc((size_t)L_ * H_ * 2);
    unsigned short* wvb  = (unsigned short*)wsAlloc((size_t)L_ * H_ * 2);
    unsigned short* wob  = (unsigned short*)wsAlloc((size_t)H_ * D_ * 2);
    unsigned short* w1b  = (unsigned short*)wsAlloc((size_t)D_ * 4 * D_ * 2);
    unsigned short* w2b  = (unsigned short*)wsAlloc((size_t)4 * D_ * D_ * 2);

    auto cvt = [&](const float* src, unsigned short* dst, int n) {
        int n4 = n / 4;
        cvt_bf16x4_kernel<<<(n4 + 255) / 256, 256, 0, stream>>>(src, dst, n4);
    };
    cvt(Wq, wqb, D_ * H_);
    cvt(Wdown, wdb, D_ * L_);
    cvt(Wk, wkb, L_ * H_);
    cvt(Wv, wvb, L_ * H_);
    cvt(Wo, wob, H_ * D_);
    cvt(W1, w1b, D_ * 4 * D_);
    cvt(W2, w2b, 4 * D_ * D_);

    // 1) h = rmsnorm(x, g1)  (bf16)
    rmsnorm_bf16_kernel<<<S_, 256, 0, stream>>>(x, g1, hb, D_);

    // 2) q = h @ Wq + bq     [S,H] bf16
    gemm_bf16_kernel<0><<<dim3(H_ / BN, S_ / BM), 256, 0, stream>>>(
        hb, wqb, bq, nullptr, qb, S_, H_, D_);
    // 3) c = h @ Wdown + bdown  [S,L] bf16
    gemm_bf16_kernel<0><<<dim3(L_ / BN, S_ / BM), 256, 0, stream>>>(
        hb, wdb, bdown, nullptr, cb, S_, L_, D_);
    // 4) k = c @ Wk + bk     [S,H] bf16
    gemm_bf16_kernel<0><<<dim3(H_ / BN, S_ / BM), 256, 0, stream>>>(
        cb, wkb, bk, nullptr, kbuf, S_, H_, L_);
    // 5) v = c @ Wv + bv     stored transposed [H,S] bf16
    gemm_bf16_kernel<1><<<dim3(H_ / BN, S_ / BM), 256, 0, stream>>>(
        cb, wvb, bv, nullptr, vtb, S_, H_, L_);

    // 6) causal flash attention -> ob [S,H] bf16
    mla_attention_kernel<<<(NH_ * (S_ / 16)) / 4, 128, 0, stream>>>(qb, kbuf, vtb, ob);

    // 7) stream = ob @ Wo + bo + x   [S,D] f32
    gemm_bf16_kernel<2><<<dim3(D_ / BN, S_ / BM), 256, 0, stream>>>(
        ob, wob, bo, x, st32, S_, D_, H_);

    // 8) h2 = rmsnorm(stream, g2)  (bf16)
    rmsnorm_bf16_kernel<<<S_, 256, 0, stream>>>(st32, g2, h2b, D_);

    // 9) f1 = silu(h2 @ W1 + b1)   [S,4D] bf16
    gemm_bf16_kernel<3><<<dim3(4 * D_ / BN, S_ / BM), 256, 0, stream>>>(
        h2b, w1b, b1, nullptr, f1b, S_, 4 * D_, D_);

    // 10) out = silu(f1 @ W2 + b2) + stream  [S,D] f32
    gemm_bf16_kernel<4><<<dim3(D_ / BN, S_ / BM), 256, 0, stream>>>(
        f1b, w2b, b2, st32, (float*)d_out, S_, D_, 4 * D_);

    (void)in_sizes; (void)n_in; (void)out_size; (void)ws_size;
}
